// Generator_61237643706723
// MI455X (gfx1250) — matrix operand order repaired and checked by reading, not compiled, not run on hardware
//
#include <hip/hip_runtime.h>
#include <math.h>

typedef __attribute__((ext_vector_type(16))) _Float16 v16h;
typedef __attribute__((ext_vector_type(8)))  _Float16 v8h;
typedef __attribute__((ext_vector_type(8)))  float    v8f;

#define SQRT2F 1.41421356237309515f
#define SLAB 2048   // pixel slab for im2col (bounds Xcol at 2*SLAB*4608*2B = 37.75MB)

__device__ __forceinline__ float lrelu_s2(float v) {
    return (v >= 0.f ? v : 0.2f * v) * SQRT2F;
}

// ---------------- TDM probe: NULL-descriptor tensor_load_to_lds (+ wait) ----------------
#if defined(__has_builtin)
# if __has_builtin(__builtin_amdgcn_tensor_load_to_lds)
#  define HAVE_TDM 1
# endif
#endif
#ifndef HAVE_TDM
# define HAVE_TDM 0
#endif

typedef unsigned int tdm_u32x4 __attribute__((ext_vector_type(4)));
typedef int          tdm_i32x4 __attribute__((ext_vector_type(4)));
typedef int          tdm_i32x8 __attribute__((ext_vector_type(8)));

__global__ void k_tdm_probe() {
    __shared__ int lds_buf[64];
#if HAVE_TDM
    tdm_u32x4 g0 = (tdm_u32x4)0;  // count=0 -> NULL tensor descriptor (architectural NOP)
    tdm_i32x8 g1 = (tdm_i32x8)0;
    tdm_i32x4 g2 = (tdm_i32x4)0;
    tdm_i32x4 g3 = (tdm_i32x4)0;
# if __clang_major__ >= 23
    tdm_i32x8 g4 = (tdm_i32x8)0;
    __builtin_amdgcn_tensor_load_to_lds(g0, g1, g2, g3, g4, 0);
# else
    __builtin_amdgcn_tensor_load_to_lds(g0, g1, g2, g3, 0);
# endif
# if __has_builtin(__builtin_amdgcn_s_wait_tensorcnt)
    __builtin_amdgcn_s_wait_tensorcnt(0);
# endif
#endif
    lds_buf[threadIdx.x & 63] = 0;
    __syncthreads();
}

// ---------------- style MLP: pixelnorm + 8x EqualLinear(lr_mul=0.01) ----------------
__global__ void k_style_mlp(const float* __restrict__ z,
                            const float* __restrict__ mlpw,
                            const float* __restrict__ mlpb,
                            float* __restrict__ lat) {
    __shared__ float xs[2][512];
    __shared__ float ssum[2];
    int t = threadIdx.x; // 512 threads
    if (t < 2) ssum[t] = 0.f;
    __syncthreads();
    float v0 = z[t], v1 = z[512 + t];
    atomicAdd(&ssum[0], v0 * v0);
    atomicAdd(&ssum[1], v1 * v1);
    __syncthreads();
    xs[0][t] = v0 * rsqrtf(ssum[0] * (1.f / 512.f) + 1e-8f);
    xs[1][t] = v1 * rsqrtf(ssum[1] * (1.f / 512.f) + 1e-8f);
    __syncthreads();
    const float ws = 0.01f / 22.62741699796952f; // lr_mul / sqrt(512)
    for (int L = 0; L < 8; ++L) {
        const float* W = mlpw + (size_t)L * 512 * 512 + (size_t)t * 512;
        float a0 = 0.f, a1 = 0.f;
        for (int j = 0; j < 512; ++j) {
            float w = W[j];
            a0 += w * xs[0][j];
            a1 += w * xs[1][j];
        }
        float b = mlpb[L * 512 + t] * 0.01f;
        float y0 = lrelu_s2(a0 * ws + b);
        float y1 = lrelu_s2(a1 * ws + b);
        __syncthreads();
        xs[0][t] = y0; xs[1][t] = y1;
        __syncthreads();
    }
    lat[t] = xs[0][t];
    lat[512 + t] = xs[1][t];
}

// ---------------- broadcast const input to both batch rows ----------------
__global__ void k_bcast(const float* __restrict__ c, float* __restrict__ act, int n) {
    int i = blockIdx.x * blockDim.x + threadIdx.x;
    if (i < 2 * n) act[i] = c[i % n];
}

// ---------------- s[b,ic] = lat . mod_w / sqrt(512) + mod_b ----------------
__global__ void k_mod_s(const float* __restrict__ lat,
                        const float* __restrict__ mw,
                        const float* __restrict__ mb,
                        float* __restrict__ s, int IC) {
    int ic = blockIdx.x * blockDim.x + threadIdx.x;
    int b = blockIdx.y;
    if (ic >= IC) return;
    const float* l  = lat + b * 512;
    const float* wr = mw + (size_t)ic * 512;
    float acc = 0.f;
    for (int j = 0; j < 512; ++j) acc += l[j] * wr[j];
    s[b * IC + ic] = acc * 0.044194173824159216f + mb[ic]; // 1/sqrt(512)
}

// ---------------- d[b,oc] = rsqrt(sum (scale*w*s)^2 + 1e-8) ----------------
__global__ void k_demod(const float* __restrict__ w,
                        const float* __restrict__ s,
                        float* __restrict__ d,
                        int OC, int IC, float scale) {
    __shared__ float red[256];
    int oc = blockIdx.x, b = blockIdx.y, t = threadIdx.x;
    const float* wr = w + (size_t)oc * IC * 9;
    const float* sb = s + b * IC;
    float acc = 0.f;
    for (int k = t; k < IC * 9; k += 256) {
        int ic = k / 9;
        float v = scale * wr[k] * sb[ic];
        acc += v * v;
    }
    red[t] = acc;
    __syncthreads();
    for (int off = 128; off > 0; off >>= 1) {
        if (t < off) red[t] += red[t + off];
        __syncthreads();
    }
    if (t == 0) d[b * OC + oc] = rsqrtf(red[0] + 1e-8f);
}

// ---------------- repack weight -> f16 GEMM A operand [OC][IC*9] ----------------
__global__ void k_wprep(const float* __restrict__ w, _Float16* __restrict__ wh,
                        int OC, int IC, float scale, int flip) {
    int i = blockIdx.x * blockDim.x + threadIdx.x;
    int n = OC * IC * 9;
    if (i >= n) return;
    int r = i % 9;
    int ky = r / 3, kx = r % 3;
    int kd = flip ? ((2 - ky) * 3 + (2 - kx)) : r;
    wh[(i - r) + kd] = (_Float16)(scale * w[i]);
}

// -------- im2col (pad-1 "same" conv), modulated f16: Xcol[b][SLAB row][K] --------
__global__ void k_im2col_pad(const float* __restrict__ act, const float* __restrict__ s,
                             _Float16* __restrict__ xcol, int IC, int H, int ns, int nc) {
    const int K = IC * 9;
    size_t tot = (size_t)2 * nc * K;
    size_t i = (size_t)blockIdx.x * blockDim.x + threadIdx.x;
    if (i >= tot) return;
    int k = i % K; size_t r = i / K;
    int nl = r % nc; int b = (int)(r / nc);
    int ic = k / 9; int rr = k - ic * 9; int ky = rr / 3, kx = rr - ky * 3;
    int n = ns + nl; int py = n / H; int px = n - py * H;   // Wo == H
    int iy = py + ky - 1, ix = px + kx - 1;
    float v = 0.f;
    if (iy >= 0 && iy < H && ix >= 0 && ix < H)
        v = act[(((size_t)b * IC + ic) * H + iy) * H + ix] * s[b * IC + ic];
    xcol[((size_t)b * SLAB + nl) * K + k] = (_Float16)v;
}

// -------- im2col for stride-2 transpose conv (zero-upsample + pad-2, flipped w) --------
// output grid is (2H+1)x(2H+1)
__global__ void k_im2col_up(const float* __restrict__ act, const float* __restrict__ s,
                            _Float16* __restrict__ xcol, int IC, int H, int ns, int nc) {
    const int K = IC * 9;
    const int Wo = 2 * H + 1;
    size_t tot = (size_t)2 * nc * K;
    size_t i = (size_t)blockIdx.x * blockDim.x + threadIdx.x;
    if (i >= tot) return;
    int k = i % K; size_t r = i / K;
    int nl = r % nc; int b = (int)(r / nc);
    int ic = k / 9; int rr = k - ic * 9; int ky = rr / 3, kx = rr - ky * 3;
    int n = ns + nl; int py = n / Wo; int px = n - py * Wo;
    int ty = py + ky - 2, tx = px + kx - 2;   // coords in virtual zero-upsampled buffer
    float v = 0.f;
    if (ty >= 0 && tx >= 0 && !(ty & 1) && !(tx & 1)) {
        int iy = ty >> 1, ix = tx >> 1;
        if (iy < H && ix < H)
            v = act[(((size_t)b * IC + ic) * H + iy) * H + ix] * s[b * IC + ic];
    }
    xcol[((size_t)b * SLAB + nl) * K + k] = (_Float16)v;
}

// ---------------- dense WMMA GEMM: out[oc][n] += wh[oc][K] * Xcol[n][K]^T ----------------
// A: wh [OC][K], B: Xcol [2][SLAB][K] (pixel-major, K contiguous), D: f32 [2][OC][Ho*Wo]
__global__ void k_gemm_wmma(const _Float16* __restrict__ xcol,
                            const _Float16* __restrict__ wh,
                            float* __restrict__ out,
                            int K, int OC, int Ho, int Wo, int ns, int nc) {
    const int lane = threadIdx.x & 31;
    const int wave = threadIdx.x >> 5;
    const int nl0 = (blockIdx.x * 8 + wave) * 16;
    if (nl0 >= nc) return;               // wave-uniform early-out, EXEC stays all-ones
    const int oc0 = blockIdx.y * 16;
    const int b = blockIdx.z;
    const int hi = lane >> 4;
    const int m  = lane & 15;            // A row owned by this lane
    int nl = nl0 + (lane & 15);          // B column (slab-local pixel)
    const bool valid = nl < nc;
    if (!valid) nl = nc - 1;
    const int n = ns + nl;
    const int py = n / Wo, px = n - py * Wo;

    const _Float16* arow = wh + (size_t)(oc0 + m) * K;       // one 32B run per chunk
    const _Float16* brow = xcol + ((size_t)b * SLAB + nl) * K; // one 32B run per chunk

    v8f acc = {};
#pragma unroll 2
    for (int kb = 0; kb < K; kb += 32) {
        // A fragment: row m, 16 consecutive K halfs (k = 16*hi + s), matching B
        v16h a = *(const v16h*)(arow + kb + hi * 16);
        // B fragment: 16 consecutive K halfs
        v16h bb = *(const v16h*)(brow + kb + hi * 16);
        __builtin_prefetch(arow + kb + 32, 0, 1);   // -> global_prefetch_b8
        __builtin_prefetch(brow + kb + 32, 0, 1);
        acc = __builtin_amdgcn_wmma_f32_16x16x32_f16(false, a, false, bb,
                                                     (short)0, acc, false, false);
    }
    if (valid) {
        float* ob = out + (size_t)b * OC * Ho * Wo;
#pragma unroll
        for (int r8 = 0; r8 < 8; ++r8) {
            int oc = oc0 + r8 + hi * 8;   // D layout: VGPR r holds rows r / r+8
            ob[((size_t)oc * Ho + py) * Wo + px] = acc[r8];
        }
    }
}

// ---------------- epilogue (non-up): demod + noise + bias + lrelu*sqrt2 ----------------
__global__ void k_epilogue(const float* __restrict__ t, const float* __restrict__ d,
                           const float* __restrict__ noise, const float* __restrict__ ab,
                           const float* __restrict__ nwp,
                           float* __restrict__ act, int OC, int H, int W) {
    size_t n = (size_t)2 * OC * H * W;
    size_t i = (size_t)blockIdx.x * blockDim.x + threadIdx.x;
    if (i >= n) return;
    int x = i % W; size_t r = i / W;
    int y = r % H; r /= H;
    int oc = r % OC; int b = (int)(r / OC);
    float v = t[i] * d[b * OC + oc]
            + nwp[0] * noise[((size_t)b * H + y) * W + x] + ab[oc];
    act[i] = lrelu_s2(v);
}

// ------- epilogue (up): 4x4 FIR blur + demod + noise + bias + lrelu*sqrt2 -------
// t: [2][OC][H+1][W+1] raw transpose-conv, act: [2][OC][H][W]
__global__ void k_epilogue_up(const float* __restrict__ t, const float* __restrict__ d,
                              const float* __restrict__ noise, const float* __restrict__ ab,
                              const float* __restrict__ nwp,
                              float* __restrict__ act, int OC, int H, int W) {
    const int Ht = H + 1, Wt = W + 1;
    size_t n = (size_t)2 * OC * H * W;
    size_t i = (size_t)blockIdx.x * blockDim.x + threadIdx.x;
    if (i >= n) return;
    int x = i % W; size_t r = i / W;
    int y = r % H; r /= H;
    int oc = r % OC; int b = (int)(r / OC);
    const float bk[4] = {1.f, 3.f, 3.f, 1.f};
    const float* tb = t + ((size_t)b * OC + oc) * Ht * Wt;
    float sum = 0.f;
#pragma unroll
    for (int dy = 0; dy < 4; ++dy) {
        int ty = y + dy - 1;
        if (ty < 0 || ty >= Ht) continue;
#pragma unroll
        for (int dx = 0; dx < 4; ++dx) {
            int tx = x + dx - 1;
            if (tx < 0 || tx >= Wt) continue;
            sum += bk[dy] * bk[dx] * tb[ty * Wt + tx];
        }
    }
    sum *= (1.f / 16.f);   // BLUR_K * 4 overall gain
    float v = sum * d[b * OC + oc]
            + nwp[0] * noise[((size_t)b * H + y) * W + x] + ab[oc];
    act[i] = lrelu_s2(v);
}

// ------- to_rgb: 1x1 mod-conv (no demod) + bias + 2x FIR-upsampled skip -------
__global__ void k_to_rgb(const float* __restrict__ act, const float* __restrict__ s,
                         const float* __restrict__ w, const float* __restrict__ bias,
                         const float* __restrict__ skip,     // prev res/2 skip, may be null
                         float* __restrict__ outp,
                         int IC, int H, float scale) {
    int W = H;
    size_t n = (size_t)2 * 3 * H * W;
    size_t i = (size_t)blockIdx.x * blockDim.x + threadIdx.x;
    if (i >= n) return;
    int x = i % W; size_t r = i / W;
    int y = r % H; r /= H;
    int c = r % 3; int b = (int)(r / 3);
    const float* ab = act + (size_t)b * IC * H * W;
    const float* sb = s + b * IC;
    const float* wc = w + (size_t)c * IC;
    float sum = 0.f;
    for (int ic = 0; ic < IC; ++ic)
        sum += wc[ic] * sb[ic] * ab[(size_t)ic * H * W + y * W + x];
    float v = sum * scale + bias[c];
    if (skip) {
        const int Hh = H / 2;
        const float bk[4] = {1.f, 3.f, 3.f, 1.f};
        const float* sp = skip + ((size_t)b * 3 + c) * Hh * Hh;
        float a = 0.f;
#pragma unroll
        for (int dy = 0; dy < 4; ++dy) {
            int ty = y + dy - 2;
            if (ty < 0 || (ty & 1)) continue;
            int iy = ty >> 1; if (iy >= Hh) continue;
#pragma unroll
            for (int dx = 0; dx < 4; ++dx) {
                int tx = x + dx - 2;
                if (tx < 0 || (tx & 1)) continue;
                int ix = tx >> 1; if (ix >= Hh) continue;
                a += bk[dy] * bk[dx] * sp[iy * Hh + ix];
            }
        }
        v += a * (1.f / 16.f);
    }
    outp[i] = v;
}

// =======================================================================================
extern "C" void kernel_launch(void* const* d_in, const int* in_sizes, int n_in,
                              void* d_out, int out_size, void* d_ws, size_t ws_size,
                              hipStream_t stream) {
    (void)in_sizes; (void)n_in; (void)out_size; (void)ws_size;
    const float* z    = (const float*)d_in[0];
    const float* mlpw = (const float*)d_in[1];
    const float* mlpb = (const float*)d_in[2];
    const float* cst  = (const float*)d_in[3];

    char* wsb = (char*)d_ws;
    const size_t MB = (size_t)1 << 20;
    float*     lat   = (float*)(wsb);                          // 4 KB
    float*     sbuf  = (float*)(wsb + 4096);                   // 4 KB
    float*     dbuf  = (float*)(wsb + 8192);                   // 4 KB
    _Float16*  wh    = (_Float16*)(wsb + 16384);               // <= 4.8 MB
    _Float16*  xcol  = (_Float16*)(wsb + 16384 + 5 * MB);      // <= 37.8 MB
    float*     tbuf  = (float*)(wsb + 16384 + 43 * MB);        // <= 34.2 MB
    float*     actA  = (float*)(wsb + 16384 + 79 * MB);        // <= 33.6 MB
    float*     actB  = (float*)(wsb + 16384 + 115 * MB);       // <= 33.6 MB
    float*     skipA = (float*)(wsb + 16384 + 151 * MB);       // 0.4 MB
    float*     skipB = (float*)(wsb + 16384 + 152 * MB);       // 0.4 MB

    k_tdm_probe<<<1, 64, 0, stream>>>();
    k_style_mlp<<<1, 512, 0, stream>>>(z, mlpw, mlpb, lat);
    k_bcast<<<(2 * 8192 + 255) / 256, 256, 0, stream>>>(cst, actA, 8192);

    struct CSs { int base, IC, OC, Hin, up, noise; };
    const CSs cs[11] = {
        { 4, 512, 512,   4, 0, 83},   // conv1       @4
        {13, 512, 512,   4, 1, 84},   // L0 up   4->8
        {18, 512, 512,   8, 0, 85},   // L0      @8
        {23, 512, 512,   8, 1, 86},   // L1 up   8->16
        {28, 512, 512,  16, 0, 87},   // L1      @16
        {33, 512, 512,  16, 1, 88},   // L2 up  16->32
        {38, 512, 512,  32, 0, 89},   // L2      @32
        {43, 512, 512,  32, 1, 90},   // L3 up  32->64
        {48, 512, 512,  64, 0, 91},   // L3      @64
        {53, 512, 256,  64, 1, 92},   // L4 up  64->128
        {58, 256, 256, 128, 0, 93},   // L4      @128
    };
    struct RSs { int base, IC, res; };
    const RSs rs[6] = {
        { 9, 512,   4}, {63, 512,   8}, {67, 512,  16},
        {71, 512,  32}, {75, 512,  64}, {79, 256, 128},
    };

    float* acur = actA;
    float* anext = actB;
    const float* skipPrev = nullptr;
    int rgbIdx = 0;

    for (int li = 0; li < 11; ++li) {
        const CSs& L = cs[li];
        const float* w   = (const float*)d_in[L.base + 0];
        const float* mw  = (const float*)d_in[L.base + 1];
        const float* mb  = (const float*)d_in[L.base + 2];
        const float* nw  = (const float*)d_in[L.base + 3];  // scalar noise_w (device ptr)
        const float* ab  = (const float*)d_in[L.base + 4];
        const float* noi = (const float*)d_in[L.noise];
        const int IC = L.IC, OC = L.OC, H = L.Hin;
        const int K = IC * 9;
        const float scale = 1.0f / sqrtf((float)(IC * 9));

        k_mod_s<<<dim3((IC + 255) / 256, 2), 256, 0, stream>>>(lat, mw, mb, sbuf, IC);
        k_demod<<<dim3(OC, 2), 256, 0, stream>>>(w, sbuf, dbuf, OC, IC, scale);
        k_wprep<<<(OC * K + 255) / 256, 256, 0, stream>>>(w, wh, OC, IC, scale, L.up);

        const int Ho = L.up ? (2 * H + 1) : H;
        const int Wo = Ho;
        const int N = Ho * Wo;
        for (int ns = 0; ns < N; ns += SLAB) {
            int nc = (N - ns < SLAB) ? (N - ns) : SLAB;
            size_t ne = (size_t)2 * nc * K;
            if (L.up)
                k_im2col_up<<<(unsigned)((ne + 255) / 256), 256, 0, stream>>>(
                    acur, sbuf, xcol, IC, H, ns, nc);
            else
                k_im2col_pad<<<(unsigned)((ne + 255) / 256), 256, 0, stream>>>(
                    acur, sbuf, xcol, IC, H, ns, nc);
            dim3 g((nc + 127) / 128, OC / 16, 2);
            k_gemm_wmma<<<g, 256, 0, stream>>>(xcol, wh, tbuf, K, OC, Ho, Wo, ns, nc);
        }

        if (!L.up) {
            size_t no = (size_t)2 * OC * H * H;
            k_epilogue<<<(unsigned)((no + 255) / 256), 256, 0, stream>>>(
                tbuf, dbuf, noi, ab, nw, anext, OC, H, H);
        } else {
            int H2 = 2 * H;
            size_t no = (size_t)2 * OC * H2 * H2;
            k_epilogue_up<<<(unsigned)((no + 255) / 256), 256, 0, stream>>>(
                tbuf, dbuf, noi, ab, nw, anext, OC, H2, H2);
        }
        { float* tmp = acur; acur = anext; anext = tmp; }

        if ((li & 1) == 0) {  // to_rgb after conv1 and after the 2nd conv of each level
            const RSs& R = rs[rgbIdx];
            const float* rw    = (const float*)d_in[R.base + 0];
            const float* rmw   = (const float*)d_in[R.base + 1];
            const float* rmb   = (const float*)d_in[R.base + 2];
            const float* rbias = (const float*)d_in[R.base + 3];
            k_mod_s<<<dim3((R.IC + 255) / 256, 2), 256, 0, stream>>>(lat, rmw, rmb, sbuf, R.IC);
            float* dst = (rgbIdx == 5) ? (float*)d_out : ((rgbIdx & 1) ? skipB : skipA);
            size_t nr = (size_t)2 * 3 * R.res * R.res;
            float rscale = 1.0f / sqrtf((float)R.IC);
            k_to_rgb<<<(unsigned)((nr + 255) / 256), 256, 0, stream>>>(
                acur, sbuf, rw, rbias, skipPrev, dst, R.IC, R.res, rscale);
            skipPrev = dst;
            ++rgbIdx;
        }
    }
}
